// DeformConv_7799660609549
// MI455X (gfx1250) — compile-verified, matrix operand order checked by reading, not run on hardware
//
#include <hip/hip_runtime.h>
#include <hip/hip_bf16.h>
#include <math.h>

#define BN 8
#define CH 64
#define OC 64
#define HH 128
#define WW 128
#define HW (HH * WW)
#define KT 9            // 3x3 taps
#define CK (CH * KT)    // 576 reduction length (reordered as r = t*64 + c)

typedef float  v2f  __attribute__((ext_vector_type(2)));
typedef float  v4f  __attribute__((ext_vector_type(4)));
typedef float  v8f  __attribute__((ext_vector_type(8)));
typedef __bf16 v16bf __attribute__((ext_vector_type(16)));
// float2 with 4B alignment: legal for global b64 in DWORD alignment mode
typedef float  f2u  __attribute__((ext_vector_type(2), aligned(4)));

// ---------------------------------------------------------------------------
// Kernel A: offset (18ch) + mask (9ch) 3x3 conv via fp32 WMMA (16x16x4).
// One block per (b,h) row; 8 waves; wave handles 16 pixels x 32 channels.
// Reduction tap-major: r = t*64 + c -> no divisions in the hot loop and
// tap geometry (coords, clamp, validity) is loop-invariant per tap.
// Staged weights are pair-interleaved [kpair][n][2] so each WMMA B operand
// is ONE aligned ds_load_b64 into a consecutive VGPR pair (no mov marshal).
// ---------------------------------------------------------------------------
__global__ __launch_bounds__(256)
void offmask_conv_kernel(const float* __restrict__ x,
                         const float* __restrict__ p_w, const float* __restrict__ p_b,
                         const float* __restrict__ m_w, const float* __restrict__ m_b,
                         float* __restrict__ offset, float* __restrict__ mask)
{
    __shared__ __align__(8) float wl[CK * 32];  // [kpair][n][2], 72KB
    const int tid = threadIdx.x;
    const int b   = blockIdx.x / HH;
    const int h   = blockIdx.x % HH;

    for (int idx = tid; idx < CK * 32; idx += 256) {
        int r = idx >> 5, n = idx & 31;
        int t = r >> 6, c = r & 63;          // tap-major: r = t*64 + c
        float v = 0.f;
        if (n < 18)      v = p_w[(n * CH + c) * KT + t];
        else if (n < 27) v = m_w[((n - 18) * CH + c) * KT + t];
        wl[((r >> 1) * 32 + n) * 2 + (r & 1)] = v;   // pair-interleaved
    }
    __syncthreads();

    const int lane  = tid & 31;
    const int wave  = tid >> 5;
    const int wbase = wave * 16;
    const int ln    = lane & 15;          // A row (pixel), B/D column (channel)
    const int khalf = (lane >> 4) * 2;    // f32 A 16x4: lanes<16 hold K0,K1; >=16 hold K2,K3

    const float bias0 = p_b[ln];
    const int   n1    = 16 + ln;
    const float bias1 = (n1 < 18) ? p_b[n1] : (n1 < 27 ? m_b[n1 - 18] : 0.f);

    v8f acc0, acc1;
    #pragma unroll
    for (int j = 0; j < 8; ++j) { acc0[j] = bias0; acc1[j] = bias1; }

    const float* xb = x + (size_t)b * CH * HW;
    const int pw = wbase + ln;            // this lane's pixel (x coordinate)

    for (int t = 0; t < KT; ++t) {
        const int y  = h  + (t / 3) - 1;           // loop-invariant per tap
        const int xx = pw + (t % 3) - 1;
        const float vf = (y >= 0 && y < HH && xx >= 0 && xx < WW) ? 1.f : 0.f;
        const int yc  = min(max(y, 0), HH - 1);
        const int xxc = min(max(xx, 0), WW - 1);
        const float* bp = xb + (size_t)khalf * HW + yc * WW + xxc;
        const int kt = t * 64;
        #pragma unroll 4
        for (int cb = 0; cb < CH; cb += 4) {
            const int k0 = kt + cb + khalf;        // always even
            v2f a;
            a[0] = bp[(cb    ) * HW] * vf;
            a[1] = bp[(cb + 1) * HW] * vf;
            v2f b0 = *(const v2f*)&wl[((k0 >> 1) * 32 + ln) * 2];       // ds_load_b64
            v2f b1 = *(const v2f*)&wl[((k0 >> 1) * 32 + 16 + ln) * 2];  // ds_load_b64
            acc0 = __builtin_amdgcn_wmma_f32_16x16x4_f32(false, a, false, b0, (short)0, acc0, false, false);
            acc1 = __builtin_amdgcn_wmma_f32_16x16x4_f32(false, a, false, b1, (short)0, acc1, false, false);
        }
    }

    // D layout: VGPR j, lanes<16 -> row j, lanes>=16 -> row j+8; column = ln.
    #pragma unroll
    for (int j = 0; j < 8; ++j) {
        int mm = j + ((lane >> 4) << 3);
        int w  = wbase + mm;
        offset[(((size_t)b * 18 + ln) * HH + h) * WW + w] = acc0[j];
        if (n1 < 18)
            offset[(((size_t)b * 18 + n1) * HH + h) * WW + w] = acc1[j];
        else if (n1 < 27)
            mask[(((size_t)b * KT + (n1 - 18)) * HH + h) * WW + w] =
                1.f / (1.f + __expf(-acc1[j]));
    }
}

// ---------------------------------------------------------------------------
// Kernel B: modulated deformable conv as implicit GEMM via bf16 WMMA.
// One block per (b,h); per-wave 16 pixels x 64 out-channels (4 N-tiles).
// Tap-major reduction: each 32-wide K-chunk sits inside ONE tap, so gather
// state (2 row bases + 4 element weights) loads once per chunk into regs.
// Bilinear corners are fetched as TWO global b64 loads (adjacent column
// pairs); all boundary clipping is pre-folded into the element weights.
// bf16 weights pre-swizzled in LDS to per-lane fragment order (one 32B read).
// Output staged through padded LDS and written back as coalesced float4.
// ---------------------------------------------------------------------------
__global__ __launch_bounds__(256)
void deform_wmma_kernel(const float* __restrict__ x,
                        const float* __restrict__ dw, const float* __restrict__ db,
                        const float* __restrict__ offset, const float* __restrict__ mask,
                        float* __restrict__ out)
{
    __shared__ __align__(32) __bf16 wsw[CK * OC];   // 72KB swizzled bf16 weights
    __shared__ int   gidx[WW * KT * 2];             // 9KB  : row0/row1 base indices
    __shared__ float gwt [WW * KT * 4];             // 18KB : w00,w01,w10,w11
    __shared__ float obuf[OC * (WW + 4)];           // 33KB, +4 pad kills bank conflicts

    const int tid = threadIdx.x;
    const int b   = blockIdx.x / HH;
    const int h   = blockIdx.x % HH;

    // Weight load + swizzle into tap-major fragment order.
    // Fragment element i <-> klocal = (i<8?i:i+8)+8*lh ; r = t*64 + c.
    for (int widx = tid; widx < OC * CK; widx += 256) {
        int o  = widx / CK, rem = widx % CK;
        int c  = rem / KT,  t   = rem % KT;         // d_w flat idx = (o*CH+c)*KT+t
        int r  = t * 64 + c;
        int kb = r >> 5, kl = r & 31;
        int aa = (kl >> 4) & 1, ss = (kl >> 3) & 1, rr = kl & 7;
        int i  = rr + (aa << 3);
        int nt = o >> 4, n = o & 15;
        int sflat = ((((kb << 2) + nt) * 2 + ss) * 16 + n) * 16 + i;
        wsw[sflat] = (__bf16)dw[widx];
    }

    // Bilinear gather table per (pixel, tap): two row-base indices + four
    // element weights (validity and the clamped-base remap pre-folded in).
    for (int e = tid; e < WW * KT; e += 256) {
        int p = e / KT, t = e % KT;
        int ki = t / 3, kj = t % 3;
        float dy = offset[(((size_t)b * 18 + 2 * t    ) * HH + h) * WW + p];
        float dx = offset[(((size_t)b * 18 + 2 * t + 1) * HH + h) * WW + p];
        float mk = mask  [(((size_t)b * KT + t        ) * HH + h) * WW + p];
        float py = (float)(h - 1 + ki) + dy;
        float px = (float)(p - 1 + kj) + dx;
        float y0f = floorf(py), x0f = floorf(px);
        float fy = py - y0f, fx = px - x0f;
        int y0 = (int)y0f, x0 = (int)x0f;

        // rows: clamp each, validity as 0/1 factor
        int   yr0 = min(max(y0, 0), HH - 1),     yr1 = min(max(y0 + 1, 0), HH - 1);
        float vy0 = (y0     >= 0 && y0     < HH) ? 1.f : 0.f;
        float vy1 = (y0 + 1 >= 0 && y0 + 1 < HH) ? 1.f : 0.f;
        // columns: one clamped base, remap the two corner weights onto the
        // two elements actually loaded at (xbase, xbase+1)
        int   xbase = min(max(x0, 0), WW - 2);
        float vx0 = (x0     >= 0 && x0     < WW) ? 1.f : 0.f;
        float vx1 = (x0 + 1 >= 0 && x0 + 1 < WW) ? 1.f : 0.f;
        float cwl = (1.f - fx) * vx0;            // corner column x0
        float cwr = fx * vx1;                    // corner column x0+1
        float we0 = (xbase     == x0) ? cwl : ((xbase     == x0 + 1) ? cwr : 0.f);
        float we1 = (xbase + 1 == x0) ? cwl : ((xbase + 1 == x0 + 1) ? cwr : 0.f);

        float rt = (1.f - fy) * vy0 * mk;        // top row factor
        float rb = fy * vy1 * mk;                // bottom row factor
        gidx[e * 2 + 0] = yr0 * WW + xbase;
        gidx[e * 2 + 1] = yr1 * WW + xbase;
        gwt [e * 4 + 0] = rt * we0;
        gwt [e * 4 + 1] = rt * we1;
        gwt [e * 4 + 2] = rb * we0;
        gwt [e * 4 + 3] = rb * we1;
    }
    __syncthreads();

    const int lane  = tid & 31;
    const int wave  = tid >> 5;
    const int wbase = wave * 16;
    const int ln    = lane & 15;
    const int lh    = lane >> 4;
    const int p     = wbase + ln;                   // A-row pixel for this lane

    v8f acc[4];
    #pragma unroll
    for (int nt = 0; nt < 4; ++nt) {
        float bias = db[nt * 16 + ln];              // lane's 8 D slots share column n
        #pragma unroll
        for (int j = 0; j < 8; ++j) acc[nt][j] = bias;
    }

    const float* xb = x + (size_t)b * CH * HW;

    for (int t = 0; t < KT; ++t) {
        const int e2 = (p * KT + t) * 2;            // gather state: once per tap
        const int e4 = (p * KT + t) * 4;
        const int   g0 = gidx[e2 + 0], g1 = gidx[e2 + 1];
        const float w00 = gwt[e4 + 0], w01 = gwt[e4 + 1],
                    w10 = gwt[e4 + 2], w11 = gwt[e4 + 3];
        if (t + 1 < KT)                              // warm next tap (global_prefetch)
            __builtin_prefetch(xb + gidx[e2 + 2], 0, 1);
        #pragma unroll
        for (int chl = 0; chl < 2; ++chl) {          // 2 chunks of 32 channels per tap
            const int kb32  = t * 2 + chl;
            const int cbase = chl * 32 + (lh << 3);
            v16bf afrag;
            #pragma unroll
            for (int i = 0; i < 16; ++i) {           // ISA 16-bit A 16x32 lane layout
                const int c = cbase + (i < 8 ? i : i + 8);
                const float* xcp = xb + (size_t)c * HW;
                f2u r0 = *(const f2u*)(xcp + g0);    // global_load_b64 (4B-aligned ok)
                f2u r1 = *(const f2u*)(xcp + g1);
                float v = w00 * r0[0] + w01 * r0[1] + w10 * r1[0] + w11 * r1[1];
                afrag[i] = (__bf16)v;
            }
            #pragma unroll
            for (int nt = 0; nt < 4; ++nt) {         // reuse gathered A across 4 N-tiles
                v16bf bfrag = *(const v16bf*)&wsw[((((kb32 << 2) + nt) * 2 + lh) * 16 + ln) * 16];
                acc[nt] = __builtin_amdgcn_wmma_f32_16x16x32_bf16(
                    false, afrag, false, bfrag, (short)0, acc[nt], false, false);
            }
        }
    }

    // Stage D tiles in LDS, then coalesced float4 writeback (out is 33.5MB,
    // the largest stream -> worth full coalescing at 23.3 TB/s).
    #pragma unroll
    for (int nt = 0; nt < 4; ++nt) {
        #pragma unroll
        for (int j = 0; j < 8; ++j) {
            int mm = j + (lh << 3);
            int o  = nt * 16 + ln;
            obuf[o * (WW + 4) + wbase + mm] = acc[nt][j];
        }
    }
    __syncthreads();
    for (int idx = tid; idx < OC * (WW / 4); idx += 256) {
        int o  = idx >> 5;                           // 32 float4 per channel row
        int wv = (idx & 31) << 2;
        v4f val = *(const v4f*)&obuf[o * (WW + 4) + wv];
        *(v4f*)&out[(((size_t)b * OC + o) * HH + h) * WW + wv] = val;
    }
}

// ---------------------------------------------------------------------------
extern "C" void kernel_launch(void* const* d_in, const int* in_sizes, int n_in,
                              void* d_out, int out_size, void* d_ws, size_t ws_size,
                              hipStream_t stream)
{
    (void)in_sizes; (void)n_in; (void)out_size; (void)ws_size;
    const float* x   = (const float*)d_in[0];
    const float* p_w = (const float*)d_in[1];
    const float* p_b = (const float*)d_in[2];
    const float* m_w = (const float*)d_in[3];
    const float* m_b = (const float*)d_in[4];
    const float* d_w = (const float*)d_in[5];
    const float* d_b = (const float*)d_in[6];

    float* out    = (float*)d_out;                          // (B,64,128,128)
    float* offset = out + (size_t)BN * OC * HH * WW;        // (B,18,128,128), 2nd output
    float* mask   = (float*)d_ws;                           // (B,9,128,128) scratch

    dim3 grid(BN * HH), block(256);
    offmask_conv_kernel<<<grid, block, 0, stream>>>(x, p_w, p_b, m_w, m_b, offset, mask);
    deform_wmma_kernel <<<grid, block, 0, stream>>>(x, d_w, d_b, offset, mask, out);
}
